// BilinearFeedForward_16527034155428
// MI455X (gfx1250) — matrix operand order repaired and checked by reading, not compiled, not run on hardware
//
#include <hip/hip_runtime.h>

// B=4, N=2048, D=1024 bilinear feed-forward, all GEMMs via v_wmma_f32_16x16x32_bf16.
// Double-buffered (ping-pong) operand fragments + multi-tile waves for L2-traffic reduction.
#define Bb 4
#define Nn 2048
#define Dd 1024
#define EPSV 1e-5f

typedef __attribute__((ext_vector_type(16))) __bf16 v16bf;
typedef __attribute__((ext_vector_type(8)))  float  v8f;

__device__ __forceinline__ unsigned short f2bf(float f) {
  // fp32 -> bf16, round-to-nearest-even
  unsigned int u = __float_as_uint(f);
  unsigned int r = u + 0x7FFFu + ((u >> 16) & 1u);
  return (unsigned short)(r >> 16);
}

union AFrag { uint4 u[2]; v16bf v; };           // 16x32 bf16 A fragment: two 16B chunks per lane
union H8    { unsigned short s[8]; uint4 u; };  // 8 bf16 packed for one 16B store

__device__ __forceinline__ v8f wmma_bf16(v16bf a, v16bf b, v8f c) {
  return __builtin_amdgcn_wmma_f32_16x16x32_bf16(false, a, false, b, (short)0, c, false, false);
}

// ---------------- conversion / packing ----------------

__global__ void cvt_x_kernel(const float* __restrict__ xr, const float* __restrict__ xi,
                             unsigned short* __restrict__ xr16, unsigned short* __restrict__ xi16,
                             int n) {
  int i = blockIdx.x * blockDim.x + threadIdx.x;
  if (i < n) { xr16[i] = f2bf(xr[i]); xi16[i] = f2bf(xi[i]); }
}

// Pack W[k][n] (DxD row-major, k = reduction dim) into this file's WMMA B lane order (same k order as A):
// block(nt,kc) = 32 lanes x 16 bf16; lane e<16 holds col nt*16+e, K=kc*32+0..15;
// lane e+16 holds same col, K=kc*32+16..31. One 32B load per fragment in the GEMM.
__global__ void pack_w_kernel(const float* __restrict__ w0, const float* __restrict__ w1,
                              const float* __restrict__ w2, const float* __restrict__ w3,
                              unsigned short* __restrict__ wp) {
  int t    = blockIdx.x * blockDim.x + threadIdx.x;  // [wi][nt][kc][lane]
  int lane = t & 31;
  int kc   = (t >> 5)  & (Dd / 32 - 1);
  int nt   = (t >> 10) & (Dd / 16 - 1);
  int wi   = t >> 16;
  const float* W = (wi == 0) ? w0 : (wi == 1) ? w1 : (wi == 2) ? w2 : w3;
  int col = nt * 16 + (lane & 15);
  int kb  = kc * 32 + ((lane >> 4) << 4);
  H8 lo, hi;
#pragma unroll
  for (int j = 0; j < 8; ++j) lo.s[j] = f2bf(W[(size_t)(kb + j) * Dd + col]);
#pragma unroll
  for (int j = 0; j < 8; ++j) hi.s[j] = f2bf(W[(size_t)(kb + 8 + j) * Dd + col]);
  size_t off = (size_t)wi * Dd * Dd + ((size_t)(nt * (Dd / 32) + kc) * 32 + lane) * 16;
  *(uint4*)(wp + off)     = lo.u;
  *(uint4*)(wp + off + 8) = hi.u;
}

__global__ void zero2_kernel(float* a, float* b, int n) {
  int i = blockIdx.x * blockDim.x + threadIdx.x;
  if (i < n) { a[i] = 0.0f; b[i] = 0.0f; }
}

__global__ void scales_kernel(const float* __restrict__ ssk, const float* __restrict__ ssv,
                              float* __restrict__ sck, float* __restrict__ scv, int n) {
  int i = blockIdx.x * blockDim.x + threadIdx.x;
  if (i < n) {
    sck[i] = 1.0f / fmaxf(sqrtf(ssk[i]), EPSV);
    scv[i] = 1.0f / fmaxf(sqrtf(ssv[i]), EPSV);
  }
}

// ---------------- GEMM 1: Qr, Qi, K, V (shared A streams, 2 M-tiles/wave) --------
// Each wave: 32 rows x 16 cols for 4 outputs => 8 accumulators, 8 WMMAs per k-chunk
// from 4 A-fragments + 4 B-fragments (32B/lane/WMMA). Ping-pong double buffering.
__global__ __launch_bounds__(128) void gemm1_kernel(
    const unsigned short* __restrict__ xr16, const unsigned short* __restrict__ xi16,
    const unsigned short* __restrict__ wp,
    unsigned short* __restrict__ q16, unsigned short* __restrict__ kt16,
    unsigned short* __restrict__ vt16, float* __restrict__ ssk, float* __restrict__ ssv) {
  const int lane = threadIdx.x & 31;
  const int wid  = threadIdx.x >> 5;
  const int m0   = (blockIdx.x * 4 + wid) * 32;  // global row over B*N
  const int nt   = blockIdx.y;
  const int n0   = nt * 16;
  const int lrow = lane & 15;
  const int lhi  = (lane >> 4) << 3;  // 0 or 8

  const unsigned short* ar0 = xr16 + (size_t)(m0 + lrow) * Dd + (lhi << 1);
  const unsigned short* ar1 = ar0 + (size_t)16 * Dd;
  const unsigned short* ai0 = xi16 + (size_t)(m0 + lrow) * Dd + (lhi << 1);
  const unsigned short* ai1 = ai0 + (size_t)16 * Dd;
  const size_t WS = (size_t)Dd * Dd;
  const unsigned short* wb = wp + (size_t)nt * (Dd / 32) * 512 + (size_t)lane * 16;

  v8f cQr0 = {}, cQr1 = {}, cQi0 = {}, cQi1 = {};
  v8f cK0  = {}, cK1  = {}, cV0  = {}, cV1  = {};

  AFrag A0[4], A1[4];   // [xr m-tile0, xr m-tile1, xi m-tile0, xi m-tile1]
  v16bf B0[4], B1[4];   // [Wqr, Wqi, Wk, Wv]

  auto loadA = [&](AFrag* A, int k) {
    A[0].u[0] = *(const uint4*)(ar0 + k); A[0].u[1] = *(const uint4*)(ar0 + k + 8);
    A[1].u[0] = *(const uint4*)(ar1 + k); A[1].u[1] = *(const uint4*)(ar1 + k + 8);
    A[2].u[0] = *(const uint4*)(ai0 + k); A[2].u[1] = *(const uint4*)(ai0 + k + 8);
    A[3].u[0] = *(const uint4*)(ai1 + k); A[3].u[1] = *(const uint4*)(ai1 + k + 8);
  };
  auto loadB = [&](v16bf* Bf, int k) {
    size_t bo = (size_t)(k >> 5) * 512;
    Bf[0] = *(const v16bf*)(wb + 0 * WS + bo);
    Bf[1] = *(const v16bf*)(wb + 1 * WS + bo);
    Bf[2] = *(const v16bf*)(wb + 2 * WS + bo);
    Bf[3] = *(const v16bf*)(wb + 3 * WS + bo);
  };
  auto mac = [&](AFrag* A, v16bf* Bf) {
    // pair WMMAs sharing the same A operand
    cQr0 = wmma_bf16(A[0].v, Bf[0], cQr0);
    cK0  = wmma_bf16(A[0].v, Bf[2], cK0);
    cQr1 = wmma_bf16(A[1].v, Bf[0], cQr1);
    cK1  = wmma_bf16(A[1].v, Bf[2], cK1);
    cQi0 = wmma_bf16(A[2].v, Bf[1], cQi0);
    cV0  = wmma_bf16(A[2].v, Bf[3], cV0);
    cQi1 = wmma_bf16(A[3].v, Bf[1], cQi1);
    cV1  = wmma_bf16(A[3].v, Bf[3], cV1);
  };

  loadA(A0, 0); loadB(B0, 0);
  for (int k = 0; k < Dd; k += 64) {
    loadA(A1, k + 32); loadB(B1, k + 32);
    __builtin_prefetch(ar0 + k + 256, 0, 1);  // global_prefetch_b8: x streams from HBM once
    __builtin_prefetch(ai0 + k + 256, 0, 1);
    mac(A0, B0);
    const int kn = (k + 64 < Dd) ? (k + 64) : 0;  // clamp keeps loads uniform & in-bounds
    loadA(A0, kn); loadB(B0, kn);
    mac(A1, B1);
  }

  // Epilogue per M-tile. C layout: float v of v8f -> row mb+lhi+v, col n0+lrow.
  const int col = n0 + lrow;
  auto epi = [&](int mb, v8f cQr, v8f cQi, v8f cK, v8f cV) {
    const int b    = mb / Nn;
    const int ninb = (mb % Nn) + lhi;
#pragma unroll
    for (int v = 0; v < 8; ++v)
      q16[(size_t)(mb + lhi + v) * Dd + col] = f2bf(cQr[v] * cQi[v]);
    H8 hk, hv; float sk = 0.0f, sv = 0.0f;
#pragma unroll
    for (int v = 0; v < 8; ++v) {
      float kv = cK[v]; sk += kv * kv; hk.s[v] = f2bf(kv);
      float vv = cV[v]; sv += vv * vv; hv.s[v] = f2bf(vv);
    }
    *(uint4*)(kt16 + (size_t)b * Dd * Nn + (size_t)col * Nn + ninb) = hk.u;  // Kt[d][n]
    *(uint4*)(vt16 + (size_t)b * Dd * Nn + (size_t)col * Nn + ninb) = hv.u;  // Vt[e][n]
    atomicAdd(&ssk[b * Dd + col], sk);
    atomicAdd(&ssv[b * Dd + col], sv);
  };
  epi(m0,      cQr0, cQi0, cK0, cV0);
  epi(m0 + 16, cQr1, cQi1, cK1, cV1);
}

// ---------------- GEMM 2: S = scaled K^T V (per batch, 2x2 tiles/wave) ----------
// A = Kt[d][n], B = Vt[e][n] (A·B^T form -> contiguous fragment loads).
// Epilogue applies rank-1 norm scaling and stores St[e][d] (transposed, bf16).
__global__ __launch_bounds__(128) void gemm2_kernel(
    const unsigned short* __restrict__ kt16, const unsigned short* __restrict__ vt16,
    const float* __restrict__ sck, const float* __restrict__ scv,
    unsigned short* __restrict__ st16) {
  const int lane = threadIdx.x & 31;
  const int wid  = threadIdx.x >> 5;
  const int b    = blockIdx.z;
  const int d0   = (blockIdx.x * 4 + wid) * 32;
  const int e0   = blockIdx.y * 32;
  const int lrow = lane & 15;
  const int lhi  = (lane >> 4) << 3;

  const unsigned short* Ap0 = kt16 + (size_t)b * Dd * Nn + (size_t)(d0 + lrow) * Nn + (lhi << 1);
  const unsigned short* Ap1 = Ap0 + (size_t)16 * Nn;
  const unsigned short* Bq0 = vt16 + (size_t)b * Dd * Nn + (size_t)(e0 + lrow) * Nn + (lhi << 1);
  const unsigned short* Bq1 = Bq0 + (size_t)16 * Nn;

  v8f c00 = {}, c01 = {}, c10 = {}, c11 = {};
  AFrag a0[2], a1[2];
  v16bf b0[2], b1[2];

  auto loadF = [&](AFrag* a, v16bf* bf, int k) {
    a[0].u[0] = *(const uint4*)(Ap0 + k); a[0].u[1] = *(const uint4*)(Ap0 + k + 8);
    a[1].u[0] = *(const uint4*)(Ap1 + k); a[1].u[1] = *(const uint4*)(Ap1 + k + 8);
    bf[0] = *(const v16bf*)(Bq0 + k);
    bf[1] = *(const v16bf*)(Bq1 + k);
  };
  auto mac = [&](AFrag* a, v16bf* bf) {
    c00 = wmma_bf16(a[0].v, bf[0], c00);
    c01 = wmma_bf16(a[0].v, bf[1], c01);
    c10 = wmma_bf16(a[1].v, bf[0], c10);
    c11 = wmma_bf16(a[1].v, bf[1], c11);
  };

  loadF(a0, b0, 0);
  for (int k = 0; k < Nn; k += 64) {
    loadF(a1, b1, k + 32);
    mac(a0, b0);
    const int kn = (k + 64 < Nn) ? (k + 64) : 0;
    loadF(a0, b0, kn);
    mac(a1, b1);
  }

  auto epi = [&](int db, int eb, v8f c) {
    const int e  = eb + lrow;
    float se = scv[b * Dd + e];
    H8 h;
#pragma unroll
    for (int v = 0; v < 8; ++v) {
      int d = db + lhi + v;
      h.s[v] = f2bf(c[v] * sck[b * Dd + d] * se);
    }
    *(uint4*)(st16 + (size_t)b * Dd * Dd + (size_t)e * Dd + db + lhi) = h.u;  // St[e][d]
  };
  epi(d0,      e0,      c00);
  epi(d0,      e0 + 16, c01);
  epi(d0 + 16, e0,      c10);
  epi(d0 + 16, e0 + 16, c11);
}

// ---------------- GEMM 3: out = query @ kv + bias (2x2 tiles/wave) ----------
__global__ __launch_bounds__(128) void gemm3_kernel(
    const unsigned short* __restrict__ q16, const unsigned short* __restrict__ st16,
    const float* __restrict__ bias, float* __restrict__ out) {
  const int lane = threadIdx.x & 31;
  const int wid  = threadIdx.x >> 5;
  const int m0   = (blockIdx.x * 4 + wid) * 32;  // global row over B*N
  const int e0   = blockIdx.y * 32;
  const int lrow = lane & 15;
  const int lhi  = (lane >> 4) << 3;
  const int b    = m0 / Nn;

  const unsigned short* Ap0 = q16 + (size_t)(m0 + lrow) * Dd + (lhi << 1);
  const unsigned short* Ap1 = Ap0 + (size_t)16 * Dd;
  const unsigned short* Bq0 = st16 + (size_t)b * Dd * Dd + (size_t)(e0 + lrow) * Dd + (lhi << 1);
  const unsigned short* Bq1 = Bq0 + (size_t)16 * Dd;

  v8f c00 = {}, c01 = {}, c10 = {}, c11 = {};
  AFrag a0[2], a1[2];
  v16bf b0[2], b1[2];

  auto loadF = [&](AFrag* a, v16bf* bf, int k) {
    a[0].u[0] = *(const uint4*)(Ap0 + k); a[0].u[1] = *(const uint4*)(Ap0 + k + 8);
    a[1].u[0] = *(const uint4*)(Ap1 + k); a[1].u[1] = *(const uint4*)(Ap1 + k + 8);
    bf[0] = *(const v16bf*)(Bq0 + k);
    bf[1] = *(const v16bf*)(Bq1 + k);
  };
  auto mac = [&](AFrag* a, v16bf* bf) {
    c00 = wmma_bf16(a[0].v, bf[0], c00);
    c01 = wmma_bf16(a[0].v, bf[1], c01);
    c10 = wmma_bf16(a[1].v, bf[0], c10);
    c11 = wmma_bf16(a[1].v, bf[1], c11);
  };

  loadF(a0, b0, 0);
  for (int k = 0; k < Dd; k += 64) {
    loadF(a1, b1, k + 32);
    mac(a0, b0);
    const int kn = (k + 64 < Dd) ? (k + 64) : 0;
    loadF(a0, b0, kn);
    mac(a1, b1);
  }

  auto epi = [&](int mb, int eb, v8f c) {
    const int e  = eb + lrow;
    float bv = bias[e];
#pragma unroll
    for (int v = 0; v < 8; ++v)
      out[(size_t)(mb + lhi + v) * Dd + e] = c[v] + bv;
  };
  epi(m0,      e0,      c00);
  epi(m0,      e0 + 16, c01);
  epi(m0 + 16, e0,      c10);
  epi(m0 + 16, e0 + 16, c11);
}

// ---------------- launch ----------------

extern "C" void kernel_launch(void* const* d_in, const int* in_sizes, int n_in,
                              void* d_out, int out_size, void* d_ws, size_t ws_size,
                              hipStream_t stream) {
  const float* xr   = (const float*)d_in[0];
  const float* xi   = (const float*)d_in[1];
  const float* wqr  = (const float*)d_in[2];
  const float* wqi  = (const float*)d_in[3];
  const float* wk   = (const float*)d_in[4];
  const float* wv   = (const float*)d_in[5];
  const float* bias = (const float*)d_in[6];
  float* out = (float*)d_out;

  char* p = (char*)d_ws;
  auto take = [&p](size_t bytes) -> char* {
    char* r = p;
    p += (bytes + 255) & ~(size_t)255;
    return r;
  };
  const size_t nx = (size_t)Bb * Nn * Dd;
  unsigned short* xr16 = (unsigned short*)take(nx * 2);
  unsigned short* xi16 = (unsigned short*)take(nx * 2);
  unsigned short* wpk  = (unsigned short*)take((size_t)4 * Dd * Dd * 2);
  unsigned short* q16  = (unsigned short*)take(nx * 2);
  unsigned short* kt16 = (unsigned short*)take(nx * 2);
  unsigned short* vt16 = (unsigned short*)take(nx * 2);
  unsigned short* st16 = (unsigned short*)take((size_t)Bb * Dd * Dd * 2);
  float* ssk = (float*)take((size_t)Bb * Dd * 4);
  float* ssv = (float*)take((size_t)Bb * Dd * 4);
  float* sck = (float*)take((size_t)Bb * Dd * 4);
  float* scv = (float*)take((size_t)Bb * Dd * 4);

  cvt_x_kernel<<<dim3((unsigned)((nx + 255) / 256)), 256, 0, stream>>>(xr, xi, xr16, xi16, (int)nx);
  pack_w_kernel<<<dim3((4 * 64 * 32 * 32) / 256), 256, 0, stream>>>(wqr, wqi, wk, wv, wpk);
  zero2_kernel<<<dim3((Bb * Dd + 255) / 256), 256, 0, stream>>>(ssk, ssv, Bb * Dd);

  dim3 g1(Bb * Nn / 128, Dd / 16);          // waves: 256 m-supertiles x 64 n-tiles
  gemm1_kernel<<<g1, 128, 0, stream>>>(xr16, xi16, wpk, q16, kt16, vt16, ssk, ssv);

  scales_kernel<<<dim3((Bb * Dd + 255) / 256), 256, 0, stream>>>(ssk, ssv, sck, scv, Bb * Dd);

  dim3 g2(Dd / 128, Dd / 32, Bb);           // waves: 32 x 32 supertiles per batch
  gemm2_kernel<<<g2, 128, 0, stream>>>(kt16, vt16, sck, scv, st16);

  dim3 g3(Bb * Nn / 128, Dd / 32);          // waves: 256 x 32 supertiles
  gemm3_kernel<<<g3, 128, 0, stream>>>(q16, st16, bias, out);
}
